// SelfAttentionBlock_18459769438419
// MI455X (gfx1250) — matrix operand order repaired and checked by reading, not compiled, not run on hardware
//
#include <hip/hip_runtime.h>
#include <hip/hip_bf16.h>

typedef __attribute__((ext_vector_type(16))) _Float16 v16h;
typedef __attribute__((ext_vector_type(8)))  float    v8f;

#define BSZ   128
#define NSEQ  256
#define DMODEL 512
#define HEADS 8
#define HDIM  64
#define HID   1365
#define MROWS (BSZ*NSEQ)     // 32768
#define EPS   1e-3f
#define NEGV  -1e9f

// ---- WMMA fragment index maps (CDNA5 ISA 7.12.2, wave32) ----
// A-matrix 16x32 f16: lanes 0-15 rows M=0..15 (K-halves 0), lanes 16-31 same rows (K-halves +8)
__device__ __forceinline__ int a_kmap(int lane, int e) {
  int khalf = (lane >> 4) & 1;
  int vg = e >> 1, pos = e & 1;
  int base = (vg < 4) ? (vg * 2 + pos) : (16 + (vg - 4) * 2 + pos);
  return base + khalf * 8;
}
// B-matrix 32x16 f16: same K order as A (lanes 0-15 K=0..7,16..23; lanes 16-31 K=8..15,24..31)
__device__ __forceinline__ int b_kmap(int lane, int e) {
  return ((lane >> 4) & 1) * 8 + ((e < 8) ? e : (e + 8));
}

// ---------------- weight convert + transpose: W[K,N] f32 -> WT[N,K] f16 ----------------
__global__ void cvtT_kernel(const float* __restrict__ W, _Float16* __restrict__ WT,
                            int K, int N) {
  long idx = (long)blockIdx.x * blockDim.x + threadIdx.x;
  if (idx >= (long)K * N) return;
  int n = (int)(idx / K);
  int k = (int)(idx % K);
  WT[idx] = (_Float16)W[(long)k * N + n];
}

// ---------------- LayerNorm rows of 512, f32 in -> f16 out (wave per row) ----------------
__global__ __launch_bounds__(256)
void ln512_kernel(const float* __restrict__ X, const float* __restrict__ g,
                  const float* __restrict__ bb, _Float16* __restrict__ out, int rows) {
  int lane = threadIdx.x & 31;
  int wid  = threadIdx.x >> 5;
  int row  = blockIdx.x * 8 + wid;
  if (row >= rows) return;
  const float* xp = X + (long)row * DMODEL;
  float vals[16];
  float s = 0.f;
#pragma unroll
  for (int t = 0; t < 16; ++t) { vals[t] = xp[lane + 32 * t]; s += vals[t]; }
#pragma unroll
  for (int o = 16; o > 0; o >>= 1) s += __shfl_xor(s, o, 32);
  float mean = s * (1.f / DMODEL);
  float vs = 0.f;
#pragma unroll
  for (int t = 0; t < 16; ++t) { float d = vals[t] - mean; vs += d * d; }
#pragma unroll
  for (int o = 16; o > 0; o >>= 1) vs += __shfl_xor(vs, o, 32);
  float inv = rsqrtf(vs * (1.f / DMODEL) + EPS);
  _Float16* op = out + (long)row * DMODEL;
#pragma unroll
  for (int t = 0; t < 16; ++t) {
    int j = lane + 32 * t;
    op[j] = (_Float16)((vals[t] - mean) * inv * g[j] + bb[j]);
  }
}

// ---------------- Interior LayerNorm rows of 1365, f16 in-place (block per row) --------
__global__ __launch_bounds__(256)
void ln_hid_kernel(_Float16* __restrict__ Hd, const float* __restrict__ g,
                   const float* __restrict__ bb) {
  __shared__ float red[256];
  int tid = threadIdx.x;
  _Float16* hp = Hd + (long)blockIdx.x * HID;
  float vals[6];
  float s = 0.f;
#pragma unroll
  for (int t = 0; t < 6; ++t) {
    int j = tid + 256 * t;
    float v = (j < HID) ? (float)hp[j] : 0.f;
    vals[t] = v; s += v;
  }
  red[tid] = s; __syncthreads();
#pragma unroll
  for (int o = 128; o > 0; o >>= 1) { if (tid < o) red[tid] += red[tid + o]; __syncthreads(); }
  float mean = red[0] * (1.f / HID);
  __syncthreads();
  float vs = 0.f;
#pragma unroll
  for (int t = 0; t < 6; ++t) {
    int j = tid + 256 * t;
    if (j < HID) { float d = vals[t] - mean; vs += d * d; }
  }
  red[tid] = vs; __syncthreads();
#pragma unroll
  for (int o = 128; o > 0; o >>= 1) { if (tid < o) red[tid] += red[tid + o]; __syncthreads(); }
  float inv = rsqrtf(red[0] * (1.f / HID) + EPS);
  __syncthreads();
#pragma unroll
  for (int t = 0; t < 6; ++t) {
    int j = tid + 256 * t;
    if (j < HID) hp[j] = (_Float16)((vals[t] - mean) * inv * g[j] + bb[j]);
  }
}

// ---------------- Generic WMMA GEMM, f16 A[M,K] x f16 WT[N,K] -> epilogue --------------
// Each wave computes a 16 x (16*NT) output tile: one A fragment amortized over NT WMMAs
// per K-step (NT=4 for single-weight modes, NT=2 for the dual-accumulator GLU mode).
// MODE 0: qkv  : +bias, scatter f16 to q/k/v [B,H,N,HD] (out16 = q base, k,v follow)
// MODE 1: oproj: outf = resid + (acc+bias)*gamma  (f32 [M,N])
// MODE 2: glu  : dual acc (W1=wide, W2=gate), out16 = gelu(aw)*ag  f16 [M,N]
// MODE 3: dense: outf = resid + acc*gamma
template <int MODE, int NT>
__global__ __launch_bounds__(256)
void gemm16_kernel(const _Float16* __restrict__ A, const _Float16* __restrict__ W1,
                   const _Float16* __restrict__ W2, const float* __restrict__ bias,
                   const float* __restrict__ resid, const float* __restrict__ gamma,
                   float* __restrict__ outf, _Float16* __restrict__ out16,
                   int M, int N, int K) {
  int lane = threadIdx.x & 31;
  int wid  = threadIdx.x >> 5;
  int n0   = (blockIdx.x * 8 + wid) * (16 * NT);
  int m0   = blockIdx.y * 16;
  if (n0 >= N) return;
  int row = m0 + (lane & 15);
  int cols[NT];
  bool colok[NT];
#pragma unroll
  for (int nt = 0; nt < NT; ++nt) {
    cols[nt] = n0 + nt * 16 + (lane & 15);
    colok[nt] = cols[nt] < N;
  }
  v8f acc1[NT];
  v8f acc2[NT];
#pragma unroll
  for (int nt = 0; nt < NT; ++nt) { acc1[nt] = (v8f){}; acc2[nt] = (v8f){}; }

  for (int k0 = 0; k0 < K; k0 += 32) {
    v16h a;
#pragma unroll
    for (int e = 0; e < 16; ++e) {
      int ka = k0 + a_kmap(lane, e);
      a[e] = (ka < K) ? A[(long)row * K + ka] : (_Float16)0.f;
    }
#pragma unroll
    for (int nt = 0; nt < NT; ++nt) {
      v16h w1, w2;
#pragma unroll
      for (int e = 0; e < 16; ++e) {
        int kb = k0 + b_kmap(lane, e);
        w1[e] = (colok[nt] && kb < K) ? W1[(long)cols[nt] * K + kb] : (_Float16)0.f;
        if (MODE == 2)
          w2[e] = (colok[nt] && kb < K) ? W2[(long)cols[nt] * K + kb] : (_Float16)0.f;
      }
      acc1[nt] = __builtin_amdgcn_wmma_f32_16x16x32_f16(false, a, false, w1, (short)0,
                                                        acc1[nt], false, false);
      if (MODE == 2)
        acc2[nt] = __builtin_amdgcn_wmma_f32_16x16x32_f16(false, a, false, w2, (short)0,
                                                          acc2[nt], false, false);
    }
  }

  int rbase = m0 + ((lane >> 4) & 1) * 8;
#pragma unroll
  for (int nt = 0; nt < NT; ++nt) {
    if (!colok[nt]) continue;
    int col = cols[nt];
#pragma unroll
    for (int r = 0; r < 8; ++r) {
      int m = rbase + r;
      float v = acc1[nt][r];
      if (MODE == 0) {
        v += bias[col];
        int sel = col >> 9;          // 0=q, 1=k, 2=v
        int h   = (col >> 6) & 7;
        int d   = col & 63;
        int b   = m >> 8;            // m / NSEQ
        int nr  = m & 255;
        long di = (long)sel * ((long)BSZ * HEADS * NSEQ * HDIM) +
                  (((long)b * HEADS + h) * NSEQ + nr) * HDIM + d;
        out16[di] = (_Float16)v;
      } else if (MODE == 1) {
        v += bias[col];
        outf[(long)m * N + col] = resid[(long)m * N + col] + v * gamma[col];
      } else if (MODE == 2) {
        float w = v;
        float gt = acc2[nt][r];
        float t = w + 0.044715f * w * w * w;
        float gl = 0.5f * w * (1.f + tanhf(0.7978845608028654f * t));
        out16[(long)m * N + col] = (_Float16)(gl * gt);
      } else { // MODE 3
        outf[(long)m * N + col] = resid[(long)m * N + col] + v * gamma[col];
      }
    }
  }
}

// ---------------- Fused talking-heads attention ----------------------------------------
// grid = (NSEQ/16, BSZ), block = 256 (8 waves, wave h handles head h).
// dynamic LDS: float sc[HEADS][16][NSEQ] = 128 KB (CDNA5 WGP has 320 KB LDS).
__global__ __launch_bounds__(256)
void attn_kernel(const _Float16* __restrict__ q, const _Float16* __restrict__ k,
                 const _Float16* __restrict__ v,
                 const float* __restrict__ w_t1, const float* __restrict__ b_t1,
                 const float* __restrict__ w_t2, const float* __restrict__ b_t2,
                 const float* __restrict__ inter, const unsigned char* __restrict__ mask,
                 _Float16* __restrict__ o) {
  extern __shared__ float sc[];                 // [8*16*NSEQ]
  __shared__ float wt1[64], wt2[64], bt1[8], bt2[8];
  int tid = threadIdx.x, lane = tid & 31, h = tid >> 5;
  int i0 = blockIdx.x * 16;
  int b  = blockIdx.y;
  if (tid < 64) { wt1[tid] = w_t1[tid]; wt2[tid] = w_t2[tid]; }
  if (tid < 8)  { bt1[tid] = b_t1[tid]; bt2[tid] = b_t2[tid]; }
  __syncthreads();

  const _Float16* qp = q + (((long)b * HEADS + h) * NSEQ) * HDIM;
  const _Float16* kp = k + (((long)b * HEADS + h) * NSEQ) * HDIM;
  const _Float16* vp = v + (((long)b * HEADS + h) * NSEQ) * HDIM;
  int arow = lane & 15;

  // --- scores = q @ k^T  (wave h -> head h, 16 rows x 256 cols, into LDS) ---
  v16h a0, a1;
#pragma unroll
  for (int e = 0; e < 16; ++e) {
    int d = a_kmap(lane, e);
    a0[e] = qp[(i0 + arow) * HDIM + d];
    a1[e] = qp[(i0 + arow) * HDIM + 32 + d];
  }
  for (int jt = 0; jt < 16; ++jt) {
    int j0 = jt * 16;
    v16h bk0, bk1;
#pragma unroll
    for (int e = 0; e < 16; ++e) {
      int d = b_kmap(lane, e);
      bk0[e] = kp[(j0 + (lane & 15)) * HDIM + d];
      bk1[e] = kp[(j0 + (lane & 15)) * HDIM + 32 + d];
    }
    v8f acc = {};
    acc = __builtin_amdgcn_wmma_f32_16x16x32_f16(false, a0, false, bk0, (short)0, acc, false, false);
    acc = __builtin_amdgcn_wmma_f32_16x16x32_f16(false, a1, false, bk1, (short)0, acc, false, false);
    int rb = ((lane >> 4) & 1) * 8;
    int cc = j0 + (lane & 15);
#pragma unroll
    for (int r = 0; r < 8; ++r)
      sc[(h * 16 + rb + r) * NSEQ + cc] = acc[r];
  }
  __syncthreads();

  // --- talking-heads 1 (scale, mix over heads) + interaction + mask, in place ---
  {
    int j = tid;   // 256 threads cover 256 cols
    const float scale = 0.125f;   // 1/sqrt(64)
    for (int i = 0; i < 16; ++i) {
      float sv[8];
#pragma unroll
      for (int hh = 0; hh < 8; ++hh) sv[hh] = sc[(hh * 16 + i) * NSEQ + j] * scale;
      bool mk = mask[((long)b * NSEQ + i0 + i) * NSEQ + j] != 0;
      long ib = (((long)b * NSEQ + i0 + i) * NSEQ + j) * HEADS;
      float ov[8];
#pragma unroll
      for (int g = 0; g < 8; ++g) {
        float acc = bt1[g];
#pragma unroll
        for (int hh = 0; hh < 8; ++hh) acc += sv[hh] * wt1[hh * 8 + g];
        acc += inter[ib + g];
        ov[g] = mk ? acc : NEGV;
      }
#pragma unroll
      for (int g = 0; g < 8; ++g) sc[(g * 16 + i) * NSEQ + j] = ov[g];
    }
  }
  __syncthreads();

  // --- softmax over j, per row (wave h -> its head's 16 rows) ---
  for (int i = 0; i < 16; ++i) {
    float* rowp = &sc[(h * 16 + i) * NSEQ];
    float vv[8];
    float mx = -3.4e38f;
#pragma unroll
    for (int t = 0; t < 8; ++t) { vv[t] = rowp[lane + 32 * t]; mx = fmaxf(mx, vv[t]); }
#pragma unroll
    for (int off = 16; off > 0; off >>= 1) mx = fmaxf(mx, __shfl_xor(mx, off, 32));
    float sum = 0.f;
#pragma unroll
    for (int t = 0; t < 8; ++t) { vv[t] = __expf(vv[t] - mx); sum += vv[t]; }
#pragma unroll
    for (int off = 16; off > 0; off >>= 1) sum += __shfl_xor(sum, off, 32);
    float inv = 1.f / sum;
#pragma unroll
    for (int t = 0; t < 8; ++t) rowp[lane + 32 * t] = vv[t] * inv;
  }
  __syncthreads();

  // --- talking-heads 2 (post-softmax head mix), in place ---
  {
    int j = tid;
    for (int i = 0; i < 16; ++i) {
      float sv[8];
#pragma unroll
      for (int hh = 0; hh < 8; ++hh) sv[hh] = sc[(hh * 16 + i) * NSEQ + j];
      float ov[8];
#pragma unroll
      for (int g = 0; g < 8; ++g) {
        float acc = bt2[g];
#pragma unroll
        for (int hh = 0; hh < 8; ++hh) acc += sv[hh] * wt2[hh * 8 + g];
        ov[g] = acc;
      }
#pragma unroll
      for (int g = 0; g < 8; ++g) sc[(g * 16 + i) * NSEQ + j] = ov[g];
    }
  }
  __syncthreads();

  // --- out = attn @ v : wave h writes o[b, i0..i0+15, h, 0..63] (f16, [B,N,H,HD]) ---
  for (int dt = 0; dt < 4; ++dt) {
    v8f acc = {};
    for (int jt = 0; jt < 8; ++jt) {
      int j0 = jt * 32;
      v16h af, bf;
#pragma unroll
      for (int e = 0; e < 16; ++e) {
        af[e] = (_Float16)sc[(h * 16 + arow) * NSEQ + j0 + a_kmap(lane, e)];
        bf[e] = vp[(j0 + b_kmap(lane, e)) * HDIM + dt * 16 + (lane & 15)];
      }
      acc = __builtin_amdgcn_wmma_f32_16x16x32_f16(false, af, false, bf, (short)0, acc, false, false);
    }
    int rb = ((lane >> 4) & 1) * 8;
    int cc = dt * 16 + (lane & 15);
#pragma unroll
    for (int r = 0; r < 8; ++r)
      o[(((long)b * NSEQ + i0 + rb + r) * HEADS + h) * HDIM + cc] = (_Float16)acc[r];
  }
}

// =======================================================================================
extern "C" void kernel_launch(void* const* d_in, const int* in_sizes, int n_in,
                              void* d_out, int out_size, void* d_ws, size_t ws_size,
                              hipStream_t stream) {
  const float*         x        = (const float*)d_in[0];
  const unsigned char* mask     = (const unsigned char*)d_in[1];
  const float*         inter    = (const float*)d_in[2];
  const float*         ln1_g    = (const float*)d_in[3];
  const float*         ln1_b    = (const float*)d_in[4];
  const float*         w_qkv    = (const float*)d_in[5];
  const float*         b_qkv    = (const float*)d_in[6];
  const float*         w_t1     = (const float*)d_in[7];
  const float*         b_t1     = (const float*)d_in[8];
  const float*         w_t2     = (const float*)d_in[9];
  const float*         b_t2     = (const float*)d_in[10];
  const float*         w_out    = (const float*)d_in[11];
  const float*         b_out    = (const float*)d_in[12];
  const float*         gamma1   = (const float*)d_in[13];
  const float*         ln2_g    = (const float*)d_in[14];
  const float*         ln2_b    = (const float*)d_in[15];
  const float*         w_wide   = (const float*)d_in[16];
  const float*         w_gate   = (const float*)d_in[17];
  const float*         ffn_ln_g = (const float*)d_in[18];
  const float*         ffn_ln_b = (const float*)d_in[19];
  const float*         w_dense  = (const float*)d_in[20];
  const float*         gamma2   = (const float*)d_in[21];
  float* outF = (float*)d_out;

  // ---- workspace carve-out (aligned) ----
  char* base = (char*)d_ws;
  size_t off = 0;
  auto carve = [&](size_t bytes) -> char* {
    char* p = base + off;
    off += (bytes + 255) & ~(size_t)255;
    return p;
  };
  _Float16* WTqkv   = (_Float16*)carve((size_t)1536 * 512 * 2);
  _Float16* WTout   = (_Float16*)carve((size_t)512 * 512 * 2);
  _Float16* WTwide  = (_Float16*)carve((size_t)HID * 512 * 2);
  _Float16* WTgate  = (_Float16*)carve((size_t)HID * 512 * 2);
  _Float16* WTdense = (_Float16*)carve((size_t)512 * HID * 2);
  _Float16* h16     = (_Float16*)carve((size_t)MROWS * DMODEL * 2);
  _Float16* qkv16   = (_Float16*)carve((size_t)3 * BSZ * HEADS * NSEQ * HDIM * 2);
  _Float16* o16     = (_Float16*)carve((size_t)MROWS * DMODEL * 2);
  float*    x1      = (float*)   carve((size_t)MROWS * DMODEL * 4);
  _Float16* q16 = qkv16;
  _Float16* k16 = qkv16 + (size_t)BSZ * HEADS * NSEQ * HDIM;
  _Float16* v16 = k16   + (size_t)BSZ * HEADS * NSEQ * HDIM;
  _Float16* hid16 = qkv16;    // reuse: attention is finished before GLU GEMM writes here

  // ---- weight conversions (f32 [K,N] -> f16 [N,K]) ----
  auto cvt = [&](const float* W, _Float16* WT, int K, int N) {
    long n = (long)K * N;
    cvtT_kernel<<<dim3((unsigned)((n + 255) / 256)), dim3(256), 0, stream>>>(W, WT, K, N);
  };
  cvt(w_qkv,   WTqkv,   512,  1536);
  cvt(w_out,   WTout,   512,  512);
  cvt(w_wide,  WTwide,  512,  HID);
  cvt(w_gate,  WTgate,  512,  HID);
  cvt(w_dense, WTdense, HID,  512);

  // ---- 1. LN1: h = LN(x) -> f16 ----
  ln512_kernel<<<dim3(MROWS / 8), dim3(256), 0, stream>>>(x, ln1_g, ln1_b, h16, MROWS);

  // ---- 2. QKV GEMM (+bias, scatter to [B,H,N,HD] f16): 96 n-tiles / (8 waves * 4) = 3 ----
  gemm16_kernel<0, 4><<<dim3(3, MROWS / 16), dim3(256), 0, stream>>>(
      h16, WTqkv, nullptr, b_qkv, nullptr, nullptr, nullptr, q16, MROWS, 1536, 512);

  // ---- 3. fused talking-heads attention -> o16 [B,N,H*HD] f16 ----
  attn_kernel<<<dim3(NSEQ / 16, BSZ), dim3(256), HEADS * 16 * NSEQ * sizeof(float), stream>>>(
      q16, k16, v16, w_t1, b_t1, w_t2, b_t2, inter, mask, o16);

  // ---- 4. out projection: x1 = x + (o @ w_out + b_out) * gamma1 (32 tiles -> grid.x=1) ----
  gemm16_kernel<1, 4><<<dim3(1, MROWS / 16), dim3(256), 0, stream>>>(
      o16, WTout, nullptr, b_out, x, gamma1, x1, nullptr, MROWS, 512, 512);

  // ---- 5. LN2: h2 = LN(x1) -> f16 ----
  ln512_kernel<<<dim3(MROWS / 8), dim3(256), 0, stream>>>(x1, ln2_g, ln2_b, h16, MROWS);

  // ---- 6. GLU: hid = gelu(h2@w_wide)*(h2@w_gate) -> f16 [M,HID]: 86 tiles/(8*2)=6 ----
  gemm16_kernel<2, 2><<<dim3(6, MROWS / 16), dim3(256), 0, stream>>>(
      h16, WTwide, WTgate, nullptr, nullptr, nullptr, nullptr, hid16, MROWS, HID, 512);

  // ---- 7. interior LN over HID (in place, f16) ----
  ln_hid_kernel<<<dim3(MROWS), dim3(256), 0, stream>>>(hid16, ffn_ln_g, ffn_ln_b);

  // ---- 8. dense: out = x1 + (hid @ w_dense) * gamma2 (32 tiles -> grid.x=1) ----
  gemm16_kernel<3, 4><<<dim3(1, MROWS / 16), dim3(256), 0, stream>>>(
      hid16, WTdense, nullptr, nullptr, x1, gamma2, outF, nullptr, MROWS, 512, HID);
}
